// TransformerBlock_16879221473682
// MI455X (gfx1250) — matrix operand order repaired and checked by reading, not compiled, not run on hardware
//
#include <hip/hip_runtime.h>
#include <math.h>
#include <stdint.h>

#define NB 2
#define SEQ 2048
#define DIM 1024
#define NH 16
#define HD 64
#define NE 4
#define TOPK 2
#define HIDN 4096
#define T_TOK (NB*SEQ)

typedef __attribute__((ext_vector_type(16))) __bf16 bf16x16;
typedef __attribute__((ext_vector_type(8)))  __bf16 bf16x8;
typedef __attribute__((ext_vector_type(8)))  float  f32x8;
typedef __attribute__((ext_vector_type(4)))  int    i32x4;

#if __has_builtin(__builtin_amdgcn_global_load_async_to_lds_b128)
#define HAVE_ASYNC 1
typedef __attribute__((address_space(1))) i32x4 gas_i32x4;
typedef __attribute__((address_space(3))) i32x4 las_i32x4;
#else
#define HAVE_ASYNC 0
#endif

// 16-byte global -> LDS copy: async (ASYNCcnt-tracked) when available, else via VGPRs.
static __device__ __forceinline__ void cp16(const __bf16* g, __bf16* l) {
#if HAVE_ASYNC
    __builtin_amdgcn_global_load_async_to_lds_b128(
        (gas_i32x4*)(uintptr_t)g,
        (las_i32x4*)(uint32_t)(uintptr_t)l,
        0, 0);
#else
    *(bf16x8*)l = *(const bf16x8*)g;
#endif
}
static __device__ __forceinline__ void async_wait0() {
#if HAVE_ASYNC
#if __has_builtin(__builtin_amdgcn_s_wait_asynccnt)
    __builtin_amdgcn_s_wait_asynccnt(0);
#else
    asm volatile("s_wait_asynccnt 0x0" ::: "memory");
#endif
#endif
}

static __device__ __forceinline__ bf16x16 cat8(bf16x8 lo, bf16x8 hi) {
    return __builtin_shufflevector(lo, hi, 0,1,2,3,4,5,6,7,8,9,10,11,12,13,14,15);
}
static __device__ __forceinline__ f32x8 vzero() {
    f32x8 z;
    #pragma unroll
    for (int i = 0; i < 8; ++i) z[i] = 0.0f;
    return z;
}
static __device__ __forceinline__ float gelu_tanh(float x) {
    const float c0 = 0.7978845608028654f, c1 = 0.044715f;
    return 0.5f * x * (1.0f + tanhf(c0 * (x + c1 * x * x * x)));
}

// ---------------- f32 -> bf16 conversion ----------------
__global__ void cvt_bf16_kernel(const float* __restrict__ src, __bf16* __restrict__ dst, int n) {
    int i = blockIdx.x * blockDim.x + threadIdx.x;
    int stride = gridDim.x * blockDim.x;
    for (; i < n; i += stride) dst[i] = (__bf16)src[i];
}

// ---------------- LayerNorm (one token per block) ----------------
__global__ __launch_bounds__(256)
void layernorm_kernel(const float* __restrict__ x, const float* __restrict__ g,
                      const float* __restrict__ b, __bf16* __restrict__ out) {
    __shared__ float rs[256], rq[256];
    const int t = blockIdx.x;
    const float* row = x + (size_t)t * DIM;
    float s = 0.f, q = 0.f;
    for (int i = threadIdx.x; i < DIM; i += 256) { float v = row[i]; s += v; q += v * v; }
    rs[threadIdx.x] = s; rq[threadIdx.x] = q;
    __syncthreads();
    for (int off = 128; off > 0; off >>= 1) {
        if (threadIdx.x < off) { rs[threadIdx.x] += rs[threadIdx.x + off]; rq[threadIdx.x] += rq[threadIdx.x + off]; }
        __syncthreads();
    }
    const float mean = rs[0] * (1.0f / DIM);
    const float var = rq[0] * (1.0f / DIM) - mean * mean;
    const float rinv = rsqrtf(var + 1e-5f);
    for (int i = threadIdx.x; i < DIM; i += 256)
        out[(size_t)t * DIM + i] = (__bf16)((row[i] - mean) * rinv * g[i] + b[i]);
}

// ---------------- Tiled bf16 WMMA GEMM: C[M,N] = A[M,K] @ B[N,K]^T (+epilogue) --------
// Double-buffered LDS: one barrier per K-step; next tile streams (async when available)
// into the spare buffer while WMMAs consume the current one.
// MODE 0: outB = bf16(v + bias)
// MODE 1: outB = bf16(gelu(v + bias))
// MODE 2: outF = v + bias + resid
// MODE 3: outF += gate[row*gs] * (v + bias)
// MODE 4: outF += v + bias
template <int MODE>
__global__ __launch_bounds__(256)
void gemm_bf16_kernel(const __bf16* __restrict__ A, const __bf16* __restrict__ Bw,
                      const float* __restrict__ bias, const float* __restrict__ resid,
                      const float* __restrict__ gate, int gs,
                      float* __restrict__ outF, __bf16* __restrict__ outB,
                      int M, int N, int K) {
    constexpr int BM = 128, BN = 128, BK = 32, LDP = BK + 8;
    __shared__ __bf16 As[2][BM][LDP];
    __shared__ __bf16 Bs[2][BN][LDP];

    const int tid = threadIdx.x;
    const int lane = tid & 31, wid = tid >> 5;
    const int waveM = wid >> 2, waveN = wid & 3;          // 2 x 4 wave grid
    const int lmod = lane & 15, half = lane >> 4;
    const int blockM = blockIdx.y * BM, blockN = blockIdx.x * BN;
    const int ldRow = tid >> 1, ldCol = (tid & 1) * 16;

    f32x8 acc[4][2];
    #pragma unroll
    for (int i = 0; i < 4; ++i)
        #pragma unroll
        for (int j = 0; j < 2; ++j) acc[i][j] = vzero();

    auto loadTile = [&](int bsel, int k0) {
        const __bf16* aSrc = A  + (size_t)(blockM + ldRow) * K + k0 + ldCol;
        const __bf16* bSrc = Bw + (size_t)(blockN + ldRow) * K + k0 + ldCol;
        cp16(aSrc,     &As[bsel][ldRow][ldCol]);
        cp16(aSrc + 8, &As[bsel][ldRow][ldCol + 8]);
        cp16(bSrc,     &Bs[bsel][ldRow][ldCol]);
        cp16(bSrc + 8, &Bs[bsel][ldRow][ldCol + 8]);
        if (k0 + BK < K) {                  // prefetch tile after next -> global_prefetch_b8
            __builtin_prefetch(aSrc + BK, 0, 1);
            __builtin_prefetch(bSrc + BK, 0, 1);
        }
    };

    loadTile(0, 0);
    int buf = 0;
    for (int k0 = 0; k0 < K; k0 += BK) {
        async_wait0();
        __syncthreads();
        if (k0 + BK < K) loadTile(buf ^ 1, k0 + BK);

        bf16x16 af[4];
        #pragma unroll
        for (int i = 0; i < 4; ++i) {       // A frag: lane holds row, K chunks {0..7,16..23}/{8..15,24..31}
            const int r = waveM * 64 + i * 16 + lmod;
            const int c0 = half * 8;
            af[i] = cat8(*(const bf16x8*)&As[buf][r][c0], *(const bf16x8*)&As[buf][r][c0 + 16]);
        }
        bf16x16 bfr[2];
        #pragma unroll
        for (int j = 0; j < 2; ++j) {       // B frag: lane holds col, K chunks {0..7,16..23}/{8..15,24..31}
            const int r = waveN * 32 + j * 16 + lmod;
            const int c0 = half * 8;
            bfr[j] = cat8(*(const bf16x8*)&Bs[buf][r][c0], *(const bf16x8*)&Bs[buf][r][c0 + 16]);
        }
        #pragma unroll
        for (int i = 0; i < 4; ++i)
            #pragma unroll
            for (int j = 0; j < 2; ++j)
                acc[i][j] = __builtin_amdgcn_wmma_f32_16x16x32_bf16(
                    false, af[i], false, bfr[j], (short)0, acc[i][j], false, false);
        buf ^= 1;
    }

    #pragma unroll
    for (int i = 0; i < 4; ++i) {
        #pragma unroll
        for (int j = 0; j < 2; ++j) {
            const int col = blockN + waveN * 32 + j * 16 + lmod;
            const float bv = bias ? bias[col] : 0.0f;
            #pragma unroll
            for (int v = 0; v < 8; ++v) {
                const int row = blockM + waveM * 64 + i * 16 + v + 8 * half;
                const size_t idx = (size_t)row * N + col;
                float val = acc[i][j][v] + bv;
                if (MODE == 0)      outB[idx] = (__bf16)val;
                else if (MODE == 1) outB[idx] = (__bf16)gelu_tanh(val);
                else if (MODE == 2) outF[idx] = val + resid[idx];
                else if (MODE == 3) outF[idx] += gate[(size_t)row * gs] * val;
                else                outF[idx] += val;
            }
        }
    }
}

// ---------------- Flash attention: 64 q-rows/block (4 waves), 128-wide K tiles --------
__global__ __launch_bounds__(128)
void attention_kernel(const __bf16* __restrict__ qkv, __bf16* __restrict__ o) {
    constexpr int KT = 128, KP = KT + 8, DP = HD + 8;
    __shared__ __bf16 Kt[KT][DP];       // k rows x d
    __shared__ __bf16 Vt[HD][KP];       // transposed: d x k
    __shared__ __bf16 Pt[4][16][KP];    // per-wave P in A-layout staging

    const int tid = threadIdx.x;
    const int lane = tid & 31, w = tid >> 5;
    const int lmod = lane & 15, half = lane >> 4;
    const int Q0 = blockIdx.x * 64;
    const int bh = blockIdx.y, b = bh / NH, h = bh % NH;
    const size_t baseTok = (size_t)b * SEQ;
    const size_t rstr = 3 * DIM;
    const int qoff = h * HD, koff = DIM + h * HD, voff = 2 * DIM + h * HD;

    bf16x16 qf[2];                       // q A-fragments, K = d in [0,32) and [32,64)
    {
        const int s = Q0 + w * 16 + lmod;
        const __bf16* qr = qkv + (baseTok + s) * rstr + qoff;
        #pragma unroll
        for (int st = 0; st < 2; ++st) {
            const int c0 = st * 32 + half * 8;
            qf[st] = cat8(*(const bf16x8*)(qr + c0), *(const bf16x8*)(qr + c0 + 16));
        }
    }

    float mprev[8], lsum[8];
    f32x8 oacc[4];
    #pragma unroll
    for (int v = 0; v < 8; ++v) { mprev[v] = -3.0e38f; lsum[v] = 0.f; }
    #pragma unroll
    for (int jd = 0; jd < 4; ++jd) oacc[jd] = vzero();

    const int nkt = (Q0 + 64 + KT - 1) / KT;
    for (int kt = 0; kt < nkt; ++kt) {
        const int Kb = kt * KT;
        {   // cooperative tile load: thread t owns k-row t
            const __bf16* kr = qkv + (baseTok + Kb + tid) * rstr + koff;
            const __bf16* vr = qkv + (baseTok + Kb + tid) * rstr + voff;
            #pragma unroll
            for (int c = 0; c < 4; ++c) {
                cp16(kr + c * 16,     &Kt[tid][c * 16]);
                cp16(kr + c * 16 + 8, &Kt[tid][c * 16 + 8]);
                bf16x8 v0 = *(const bf16x8*)(vr + c * 16);
                bf16x8 v1 = *(const bf16x8*)(vr + c * 16 + 8);
                #pragma unroll
                for (int u = 0; u < 8; ++u) {
                    Vt[c * 16 + u][tid]     = v0[u];
                    Vt[c * 16 + 8 + u][tid] = v1[u];
                }
            }
        }
        async_wait0();
        __syncthreads();

        // scores: 8 fragments of 16x16 covering 128 k positions
        f32x8 sc[8];
        #pragma unroll
        for (int j = 0; j < 8; ++j) {
            const int r = j * 16 + lmod;
            sc[j] = vzero();
            #pragma unroll
            for (int st = 0; st < 2; ++st) {
                const int c0 = st * 32 + half * 8;
                bf16x16 kb = cat8(*(const bf16x8*)&Kt[r][c0], *(const bf16x8*)&Kt[r][c0 + 16]);
                sc[j] = __builtin_amdgcn_wmma_f32_16x16x32_bf16(
                    false, qf[st], false, kb, (short)0, sc[j], false, false);
            }
        }
        // scale + causal mask
        const int qbase = Q0 + w * 16 + 8 * half;
        #pragma unroll
        for (int j = 0; j < 8; ++j) {
            const int kg = Kb + j * 16 + lmod;
            #pragma unroll
            for (int v = 0; v < 8; ++v) {
                float s = sc[j][v] * 0.125f;
                if (kg > qbase + v) s = -1.0e9f;
                sc[j][v] = s;
            }
        }
        // online softmax: row stats live in lane-halves matching C-layout rows
        float rsum[8], mnew[8], alpha[8];
        #pragma unroll
        for (int v = 0; v < 8; ++v) {
            float m = sc[0][v];
            #pragma unroll
            for (int j = 1; j < 8; ++j) m = fmaxf(m, sc[j][v]);
            for (int msk = 1; msk < 16; msk <<= 1) m = fmaxf(m, __shfl_xor(m, msk, 32));
            mnew[v] = fmaxf(mprev[v], m);
            alpha[v] = __expf(mprev[v] - mnew[v]);
            rsum[v] = 0.f;
        }
        #pragma unroll
        for (int j = 0; j < 8; ++j)
            #pragma unroll
            for (int v = 0; v < 8; ++v) {
                float p = __expf(sc[j][v] - mnew[v]);
                sc[j][v] = p;
                rsum[v] += p;
            }
        #pragma unroll
        for (int v = 0; v < 8; ++v) {
            float r = rsum[v];
            for (int msk = 1; msk < 16; msk <<= 1) r += __shfl_xor(r, msk, 32);
            lsum[v] = lsum[v] * alpha[v] + r;
            mprev[v] = mnew[v];
        }
        #pragma unroll
        for (int jd = 0; jd < 4; ++jd)
            #pragma unroll
            for (int v = 0; v < 8; ++v) oacc[jd][v] *= alpha[v];

        // stage P (bf16) into A-layout via LDS
        #pragma unroll
        for (int j = 0; j < 8; ++j)
            #pragma unroll
            for (int v = 0; v < 8; ++v)
                Pt[w][v + 8 * half][j * 16 + lmod] = (__bf16)sc[j][v];
        __syncthreads();

        // O += P @ V  (4 d-fragments, contraction over 128 k = 4 WMMA steps)
        #pragma unroll
        for (int jd = 0; jd < 4; ++jd) {
            const int dr = jd * 16 + lmod;
            #pragma unroll
            for (int ks = 0; ks < 4; ++ks) {
                const int ca = ks * 32 + half * 8;
                bf16x16 pa = cat8(*(const bf16x8*)&Pt[w][lmod][ca],
                                  *(const bf16x8*)&Pt[w][lmod][ca + 16]);
                const int cb = ks * 32 + half * 8;
                bf16x16 vb = cat8(*(const bf16x8*)&Vt[dr][cb],
                                  *(const bf16x8*)&Vt[dr][cb + 16]);
                oacc[jd] = __builtin_amdgcn_wmma_f32_16x16x32_bf16(
                    false, pa, false, vb, (short)0, oacc[jd], false, false);
            }
        }
        __syncthreads();
    }

    #pragma unroll
    for (int jd = 0; jd < 4; ++jd)
        #pragma unroll
        for (int v = 0; v < 8; ++v) {
            const int s = Q0 + w * 16 + v + 8 * half;
            const int d = jd * 16 + lmod;
            o[(baseTok + s) * DIM + h * HD + d] = (__bf16)(oacc[jd][v] / lsum[v]);
        }
}

// ---------------- Router: softmax over E=4, top-2, aux accumulators ----------------
__global__ __launch_bounds__(256)
void gating_kernel(const __bf16* __restrict__ h2, const float* __restrict__ Wg,
                   float* __restrict__ gate, float* __restrict__ accum) {
    const int t = blockIdx.x * 8 + (threadIdx.x >> 5);
    const int lane = threadIdx.x & 31;
    const __bf16* hr = h2 + (size_t)t * DIM;
    float lg[NE];
    #pragma unroll
    for (int e = 0; e < NE; ++e) {
        float s = 0.f;
        for (int d = lane; d < DIM; d += 32) s += (float)hr[d] * Wg[e * DIM + d];
        for (int msk = 1; msk < 32; msk <<= 1) s += __shfl_xor(s, msk, 32);
        lg[e] = s;
    }
    float mx = lg[0];
    #pragma unroll
    for (int e = 1; e < NE; ++e) mx = fmaxf(mx, lg[e]);
    float p[NE], ps = 0.f;
    #pragma unroll
    for (int e = 0; e < NE; ++e) { p[e] = __expf(lg[e] - mx); ps += p[e]; }
    #pragma unroll
    for (int e = 0; e < NE; ++e) p[e] /= ps;
    int i1 = 0;
    #pragma unroll
    for (int e = 1; e < NE; ++e) if (p[e] > p[i1]) i1 = e;
    int i2 = (i1 == 0) ? 1 : 0;
    #pragma unroll
    for (int e = 0; e < NE; ++e) if (e != i1 && p[e] > p[i2]) i2 = e;
    const float wsum = p[i1] + p[i2];
    if (lane == 0) {
        #pragma unroll
        for (int e = 0; e < NE; ++e) {
            float gw = (e == i1) ? p[i1] / wsum : ((e == i2) ? p[i2] / wsum : 0.f);
            gate[(size_t)t * NE + e] = gw;
            atomicAdd(&accum[e], p[e]);
        }
        atomicAdd(&accum[NE + i1], 1.0f);
        atomicAdd(&accum[NE + i2], 1.0f);
    }
}

__global__ void aux_kernel(const float* __restrict__ accum, float* __restrict__ out0) {
    if (threadIdx.x == 0 && blockIdx.x == 0) {
        float s = 0.f;
        for (int e = 0; e < NE; ++e) {
            float fi = accum[NE + e] / ((float)T_TOK * (float)TOPK);
            float Pi = accum[e] / (float)T_TOK;
            s += fi * Pi;
        }
        out0[0] = 0.01f * (float)NE * s;
    }
}

__global__ void add_kernel(const float* __restrict__ a, const float* __restrict__ b,
                           float* __restrict__ out, int n) {
    int i = blockIdx.x * blockDim.x + threadIdx.x;
    int stride = gridDim.x * blockDim.x;
    for (; i < n; i += stride) out[i] = a[i] + b[i];
}

// ---------------- Orchestration ----------------
extern "C" void kernel_launch(void* const* d_in, const int* in_sizes, int n_in,
                              void* d_out, int out_size, void* d_ws, size_t ws_size,
                              hipStream_t stream) {
    (void)in_sizes; (void)n_in; (void)out_size; (void)ws_size;
    const float* x      = (const float*)d_in[0];
    const float* ln1g   = (const float*)d_in[1];
    const float* ln1b   = (const float*)d_in[2];
    const float* Wqkv   = (const float*)d_in[3];
    const float* bqkv   = (const float*)d_in[4];
    const float* Wo     = (const float*)d_in[5];
    const float* bo     = (const float*)d_in[6];
    const float* ln2g   = (const float*)d_in[7];
    const float* ln2b   = (const float*)d_in[8];
    const float* Wg     = (const float*)d_in[9];
    const float* We_up  = (const float*)d_in[10];
    const float* be_up  = (const float*)d_in[11];
    const float* We_dn  = (const float*)d_in[12];
    const float* be_dn  = (const float*)d_in[13];
    const float* Ws_up  = (const float*)d_in[14];
    const float* bs_up  = (const float*)d_in[15];
    const float* Ws_dn  = (const float*)d_in[16];
    const float* bs_dn  = (const float*)d_in[17];
    float* out = (float*)d_out;

    char* w = (char*)d_ws; size_t off = 0;
    auto alloc = [&](size_t bytes) { void* p = w + off; off += (bytes + 255) & ~(size_t)255; return p; };
    __bf16* wqkv_b = (__bf16*)alloc((size_t)3 * DIM * DIM * 2);
    __bf16* wo_b   = (__bf16*)alloc((size_t)DIM * DIM * 2);
    __bf16* weup_b = (__bf16*)alloc((size_t)NE * HIDN * DIM * 2);
    __bf16* wedn_b = (__bf16*)alloc((size_t)NE * DIM * HIDN * 2);
    __bf16* wsup_b = (__bf16*)alloc((size_t)HIDN * DIM * 2);
    __bf16* wsdn_b = (__bf16*)alloc((size_t)DIM * HIDN * 2);
    __bf16* h1     = (__bf16*)alloc((size_t)T_TOK * DIM * 2);
    __bf16* qkvb   = (__bf16*)alloc((size_t)T_TOK * 3 * DIM * 2);
    __bf16* obuf   = (__bf16*)alloc((size_t)T_TOK * DIM * 2);
    float*  xattn  = (float*) alloc((size_t)T_TOK * DIM * 4);
    __bf16* h2     = (__bf16*)alloc((size_t)T_TOK * DIM * 2);
    __bf16* upbuf  = (__bf16*)alloc((size_t)T_TOK * HIDN * 2);
    float*  ybuf   = (float*) alloc((size_t)T_TOK * DIM * 4);
    float*  gate   = (float*) alloc((size_t)T_TOK * NE * 4);
    float*  accum  = (float*) alloc(8 * 4);

    (void)hipMemsetAsync(ybuf, 0, (size_t)T_TOK * DIM * 4, stream);
    (void)hipMemsetAsync(accum, 0, 8 * 4, stream);

    cvt_bf16_kernel<<<2048, 256, 0, stream>>>(Wqkv, wqkv_b, 3 * DIM * DIM);
    cvt_bf16_kernel<<<2048, 256, 0, stream>>>(Wo, wo_b, DIM * DIM);
    cvt_bf16_kernel<<<4096, 256, 0, stream>>>(We_up, weup_b, NE * HIDN * DIM);
    cvt_bf16_kernel<<<4096, 256, 0, stream>>>(We_dn, wedn_b, NE * DIM * HIDN);
    cvt_bf16_kernel<<<2048, 256, 0, stream>>>(Ws_up, wsup_b, HIDN * DIM);
    cvt_bf16_kernel<<<2048, 256, 0, stream>>>(Ws_dn, wsdn_b, DIM * HIDN);

    layernorm_kernel<<<T_TOK, 256, 0, stream>>>(x, ln1g, ln1b, h1);

    dim3 gqkv(3 * DIM / 128, T_TOK / 128);
    gemm_bf16_kernel<0><<<gqkv, 256, 0, stream>>>(h1, wqkv_b, bqkv, nullptr, nullptr, 0,
                                                  nullptr, qkvb, T_TOK, 3 * DIM, DIM);

    dim3 gattn(SEQ / 64, NB * NH);
    attention_kernel<<<gattn, 128, 0, stream>>>(qkvb, obuf);

    dim3 gproj(DIM / 128, T_TOK / 128);
    gemm_bf16_kernel<2><<<gproj, 256, 0, stream>>>(obuf, wo_b, bo, x, nullptr, 0,
                                                   xattn, nullptr, T_TOK, DIM, DIM);

    layernorm_kernel<<<T_TOK, 256, 0, stream>>>(xattn, ln2g, ln2b, h2);
    gating_kernel<<<T_TOK / 8, 256, 0, stream>>>(h2, Wg, gate, accum);

    dim3 gup(HIDN / 128, T_TOK / 128);
    for (int e = 0; e < NE; ++e) {
        gemm_bf16_kernel<1><<<gup, 256, 0, stream>>>(h2, weup_b + (size_t)e * HIDN * DIM,
                                                     be_up + (size_t)e * HIDN, nullptr, nullptr, 0,
                                                     nullptr, upbuf, T_TOK, HIDN, DIM);
        gemm_bf16_kernel<3><<<gproj, 256, 0, stream>>>(upbuf, wedn_b + (size_t)e * DIM * HIDN,
                                                       be_dn + (size_t)e * DIM, nullptr, gate + e, NE,
                                                       ybuf, nullptr, T_TOK, DIM, HIDN);
    }
    gemm_bf16_kernel<1><<<gup, 256, 0, stream>>>(h2, wsup_b, bs_up, nullptr, nullptr, 0,
                                                 nullptr, upbuf, T_TOK, HIDN, DIM);
    gemm_bf16_kernel<4><<<gproj, 256, 0, stream>>>(upbuf, wsdn_b, bs_dn, nullptr, nullptr, 0,
                                                   ybuf, nullptr, T_TOK, DIM, HIDN);

    aux_kernel<<<1, 32, 0, stream>>>(accum, out);
    add_kernel<<<4096, 256, 0, stream>>>(xattn, ybuf, out + 1, T_TOK * DIM);
}
